// EncDecWithAttn_4217657884788
// MI455X (gfx1250) — compile-verified, matrix operand order checked by reading, not run on hardware
//
#include <hip/hip_runtime.h>
#include <hip/hip_bf16.h>
#include <stdint.h>

// ---------------------------------------------------------------------------
// Seq2seq GRU encoder/decoder with attention, restructured for MI455X:
//  * all GEMMs via v_wmma_f32_16x16x32_bf16, register-blocked 32x64 per wave
//    (2 A frags + 4 B frags -> 8 WMMAs per 32-K step: 1.5 b128 loads/WMMA)
//  * loop-invariant projections hoisted (enc @ W1a^T, x @ Wih^T)
//  * bf16 weight copies built once; out_W (65MB bf16) stays L2-resident
// ---------------------------------------------------------------------------

#define S_  64
#define B_  32
#define V_  32000
#define E_  512
#define H_  1024
#define H2_ 2048
#define H3_ 3072

typedef __attribute__((ext_vector_type(16))) __bf16 bf16x16;
typedef __attribute__((ext_vector_type(8)))  __bf16 bf16x8;
typedef __attribute__((ext_vector_type(8)))  float  f32x8;

__device__ __forceinline__ unsigned short f32_to_bf16(float f) {
    unsigned int u = __float_as_uint(f);
    unsigned int r = (u + 0x7FFFu + ((u >> 16) & 1u)) >> 16;
    return (unsigned short)r;
}

// Load one 16x32 bf16 WMMA fragment (two 16B chunks per lane).
__device__ __forceinline__ bf16x16 load_frag(const unsigned short* __restrict__ p) {
    bf16x8 lo = *reinterpret_cast<const bf16x8*>(p);
    bf16x8 hi = *reinterpret_cast<const bf16x8*>(p + 16);
    bf16x16 v;
#pragma unroll
    for (int i = 0; i < 8; ++i) { v[i] = lo[i]; v[8 + i] = hi[i]; }
    return v;
}

// --------------------------- conversion / setup ----------------------------

__global__ void k_f32_to_bf16(const float* __restrict__ src,
                              unsigned short* __restrict__ dst, long n) {
    long i = (long)blockIdx.x * blockDim.x + threadIdx.x;
    if (i >= n) return;
    dst[i] = f32_to_bf16(src[i]);
}

// gather embeddings -> bf16, E=512
__global__ void k_embed(const float* __restrict__ emb, const int* __restrict__ ids,
                        unsigned short* __restrict__ dst, long total) {
    long i = (long)blockIdx.x * blockDim.x + threadIdx.x;
    if (i >= total) return;
    long slot = i >> 9;            // / E
    int  j    = (int)(i & 511);
    dst[i] = f32_to_bf16(emb[(long)ids[slot] * E_ + j]);
}

// broadcast a length-H row to [B,H] in f32 + bf16
__global__ void k_bcast(const float* __restrict__ src, float* __restrict__ hf,
                        unsigned short* __restrict__ hb, int n) {
    int i = blockIdx.x * blockDim.x + threadIdx.x;
    if (i >= n) return;
    float v = src[i & (H_ - 1)];
    hf[i] = v;
    hb[i] = f32_to_bf16(v);
}

// ------------------------------- WMMA GEMM ---------------------------------
// out[M,N](f32,ldo) = A[M,K](bf16,lda) * W[N,K]^T(bf16,ldw) + bias[N]
// One wave computes a 32x64 block (2 M-tiles x 4 N-tiles -> 8 accumulators).
// Requires: M % 32 == 0, N % 64 == 0, K % 32 == 0.
__global__ void k_gemm_bf16(const unsigned short* __restrict__ A, int lda,
                            const unsigned short* __restrict__ W, int ldw,
                            const float* __restrict__ bias,
                            float* __restrict__ out, int ldo,
                            int M, int N, int K) {
    const int lane  = threadIdx.x & 31;
    const int wave  = blockIdx.x * (blockDim.x >> 5) + (threadIdx.x >> 5);
    const int nb    = N >> 6;                  // 64-column blocks
    const int total = (M >> 5) * nb;           // 32-row blocks
    if (wave >= total) return;                 // wave-uniform: EXEC stays full
    const int mb  = wave / nb;
    const int nbk = wave - mb * nb;

    const int  r15   = lane & 15;
    const int  khalf = lane >> 4;
    const long arow0 = (long)(mb * 32 + r15) * lda;
    const long arow1 = arow0 + (long)16 * lda;
    long wrow[4];
#pragma unroll
    for (int j = 0; j < 4; ++j)
        wrow[j] = (long)(nbk * 64 + j * 16 + r15) * ldw;

    f32x8 zero = {0.f, 0.f, 0.f, 0.f, 0.f, 0.f, 0.f, 0.f};
    f32x8 acc[8];                              // acc[mi*4 + nj]
#pragma unroll
    for (int i = 0; i < 8; ++i) acc[i] = zero;

    for (int k0 = 0; k0 < K; k0 += 32) {
        const int kb = k0 + khalf * 8;
        bf16x16 a0 = load_frag(A + arow0 + kb);
        bf16x16 a1 = load_frag(A + arow1 + kb);
        bf16x16 b0 = load_frag(W + wrow[0] + kb);
        bf16x16 b1 = load_frag(W + wrow[1] + kb);
        bf16x16 b2 = load_frag(W + wrow[2] + kb);
        bf16x16 b3 = load_frag(W + wrow[3] + kb);
        acc[0] = __builtin_amdgcn_wmma_f32_16x16x32_bf16(false, a0, false, b0, (short)0, acc[0], false, false);
        acc[1] = __builtin_amdgcn_wmma_f32_16x16x32_bf16(false, a0, false, b1, (short)0, acc[1], false, false);
        acc[2] = __builtin_amdgcn_wmma_f32_16x16x32_bf16(false, a0, false, b2, (short)0, acc[2], false, false);
        acc[3] = __builtin_amdgcn_wmma_f32_16x16x32_bf16(false, a0, false, b3, (short)0, acc[3], false, false);
        acc[4] = __builtin_amdgcn_wmma_f32_16x16x32_bf16(false, a1, false, b0, (short)0, acc[4], false, false);
        acc[5] = __builtin_amdgcn_wmma_f32_16x16x32_bf16(false, a1, false, b1, (short)0, acc[5], false, false);
        acc[6] = __builtin_amdgcn_wmma_f32_16x16x32_bf16(false, a1, false, b2, (short)0, acc[6], false, false);
        acc[7] = __builtin_amdgcn_wmma_f32_16x16x32_bf16(false, a1, false, b3, (short)0, acc[7], false, false);
    }

    // C/D layout: lanes 0-15 hold N=lane (M = m0+r), lanes 16-31 M base +8.
#pragma unroll
    for (int mi = 0; mi < 2; ++mi) {
        const int m0 = mb * 32 + mi * 16 + khalf * 8;
#pragma unroll
        for (int nj = 0; nj < 4; ++nj) {
            const int   n  = nbk * 64 + nj * 16 + r15;
            const float bv = bias ? bias[n] : 0.0f;
            const f32x8 a  = acc[mi * 4 + nj];
#pragma unroll
            for (int r = 0; r < 8; ++r)
                out[(long)(m0 + r) * ldo + n] = a[r] + bv;
        }
    }
}

// ------------------------------ GRU gates ----------------------------------
// gi/gh: [B,3H] (biases already folded in). Updates h (f32+bf16), optionally
// writes the new hidden state into the encoder buffer (pre-offset pointers,
// row stride 2H).
__global__ void k_gru_gate(const float* __restrict__ gi, const float* __restrict__ gh,
                           float* __restrict__ h, unsigned short* __restrict__ hbf,
                           float* __restrict__ encf, unsigned short* __restrict__ encbf,
                           int n) {
    int i = blockIdx.x * blockDim.x + threadIdx.x;
    if (i >= n) return;
    int  b    = i >> 10;
    int  j    = i & (H_ - 1);
    long base = (long)b * H3_;
    float ir = gi[base + j], iz = gi[base + H_ + j], inn = gi[base + H2_ + j];
    float hr = gh[base + j], hz = gh[base + H_ + j], hn  = gh[base + H2_ + j];
    float r  = 1.0f / (1.0f + __expf(-(ir + hr)));
    float z  = 1.0f / (1.0f + __expf(-(iz + hz)));
    float nn = tanhf(inn + r * hn);
    float hv = h[i];
    float hnew = (1.0f - z) * nn + z * hv;
    h[i]   = hnew;
    hbf[i] = f32_to_bf16(hnew);
    if (encf) {
        long e = (long)b * H2_ + j;
        encf[e]  = hnew;
        encbf[e] = f32_to_bf16(hnew);
    }
}

// ------------------------------ attention ----------------------------------
// scores[s*B+b] = sum_j w2[j]*tanh(EncProj[s*B+b,j] + HProj[b,j]); one wave each
__global__ void k_scores(const float* __restrict__ encProj, const float* __restrict__ hProj,
                         const float* __restrict__ w2, float* __restrict__ scores) {
    int wave = blockIdx.x * (blockDim.x >> 5) + (threadIdx.x >> 5);
    int lane = threadIdx.x & 31;
    if (wave >= S_ * B_) return;
    int b = wave & (B_ - 1);
    const float* ep = encProj + (long)wave * H_;
    const float* hp = hProj   + (long)b    * H_;
    float sum = 0.f;
    for (int j = lane; j < H_; j += 32) sum += w2[j] * tanhf(ep[j] + hp[j]);
#pragma unroll
    for (int m = 16; m; m >>= 1) sum += __shfl_xor(sum, m, 32);
    if (lane == 0) scores[wave] = sum;
}

// softmax over the BATCH dim (axis=1 of [S,B,1]); 32 lanes == 32 batch rows
__global__ void k_softmax_b(const float* __restrict__ scores, float* __restrict__ w) {
    int s = blockIdx.x, lane = threadIdx.x;
    float v  = scores[s * B_ + lane];
    float mx = v;
#pragma unroll
    for (int m = 16; m; m >>= 1) mx = fmaxf(mx, __shfl_xor(mx, m, 32));
    float e   = __expf(v - mx);
    float sum = e;
#pragma unroll
    for (int m = 16; m; m >>= 1) sum += __shfl_xor(sum, m, 32);
    w[s * B_ + lane] = e / sum;
}

// glimpse[b,c] = sum_s enc[s,b,c]*w[s,b]; written bf16 into dec_in slot
__global__ void k_glimpse(const float* __restrict__ encf, const float* __restrict__ w,
                          unsigned short* __restrict__ decin) {
    int i = blockIdx.x * blockDim.x + threadIdx.x;   // B * 2H
    if (i >= B_ * H2_) return;
    int b = i >> 11, c = i & (H2_ - 1);
    float s = 0.f;
    for (int t = 0; t < S_; ++t)
        s += encf[((long)(t * B_ + b) << 11) + c] * w[t * B_ + b];
    decin[(long)b * H3_ + E_ + c] = f32_to_bf16(s);
}

// dec_in[:,0:E) = tok; dec_in[:,E+2H:) = last_out (bcast row or per-row)
__global__ void k_decin_tok(const unsigned short* __restrict__ tok,
                            const unsigned short* __restrict__ last, int lastBcast,
                            unsigned short* __restrict__ decin) {
    int i = blockIdx.x * blockDim.x + threadIdx.x;   // B * E
    if (i >= B_ * E_) return;
    int b = i >> 9, j = i & (E_ - 1);
    decin[(long)b * H3_ + j]              = tok[(long)b * E_ + j];
    decin[(long)b * H3_ + E_ + H2_ + j]   = lastBcast ? last[j] : last[(long)b * E_ + j];
}

// softmax over vocab, one block per batch row
__global__ void k_softmax_v(const float* __restrict__ logits, float* __restrict__ out) {
    __shared__ float red[256];
    int b = blockIdx.x, tid = threadIdx.x;
    const float* lr = logits + (long)b * V_;
    float mx = -1e30f;
    for (int j = tid; j < V_; j += 256) mx = fmaxf(mx, lr[j]);
    red[tid] = mx; __syncthreads();
    for (int s = 128; s; s >>= 1) { if (tid < s) red[tid] = fmaxf(red[tid], red[tid + s]); __syncthreads(); }
    mx = red[0]; __syncthreads();
    float sum = 0.f;
    for (int j = tid; j < V_; j += 256) sum += __expf(lr[j] - mx);
    red[tid] = sum; __syncthreads();
    for (int s = 128; s; s >>= 1) { if (tid < s) red[tid] += red[tid + s]; __syncthreads(); }
    float inv = 1.0f / red[0];
    float* orow = out + (long)b * V_;
    for (int j = tid; j < V_; j += 256) orow[j] = __expf(lr[j] - mx) * inv;
}

// ---------------------------------------------------------------------------

extern "C" void kernel_launch(void* const* d_in, const int* in_sizes, int n_in,
                              void* d_out, int out_size, void* d_ws, size_t ws_size,
                              hipStream_t stream) {
    const int*   input_seq  = (const int*)  d_in[0];
    const int*   target_seq = (const int*)  d_in[1];
    const float* emb        = (const float*)d_in[2];
    const float* encWihF    = (const float*)d_in[3];
    const float* encWhhF    = (const float*)d_in[4];
    const float* encBihF    = (const float*)d_in[5];
    const float* encBhhF    = (const float*)d_in[6];
    const float* encWihB    = (const float*)d_in[7];
    const float* encWhhB    = (const float*)d_in[8];
    const float* encBihB    = (const float*)d_in[9];
    const float* encBhhB    = (const float*)d_in[10];
    const float* encInit    = (const float*)d_in[11];
    const float* decWih     = (const float*)d_in[12];
    const float* decWhh     = (const float*)d_in[13];
    const float* decBih     = (const float*)d_in[14];
    const float* decBhh     = (const float*)d_in[15];
    const float* decInitH   = (const float*)d_in[16];
    const float* decInitOut = (const float*)d_in[17];
    const float* attnW1     = (const float*)d_in[18];
    const float* attnW2     = (const float*)d_in[19];
    const float* outW       = (const float*)d_in[20];
    const float* outB       = (const float*)d_in[21];
    float* out = (float*)d_out;

    uint8_t* ws = (uint8_t*)d_ws;
    size_t off = 0;
    auto alloc = [&](size_t bytes) -> void* {
        off = (off + 255) & ~(size_t)255;
        void* p = ws + off;
        off += bytes;
        return p;
    };

    // bf16 weight copies
    unsigned short* wWihF   = (unsigned short*)alloc((size_t)H3_ * E_  * 2);
    unsigned short* wWhhF   = (unsigned short*)alloc((size_t)H3_ * H_  * 2);
    unsigned short* wWihB   = (unsigned short*)alloc((size_t)H3_ * E_  * 2);
    unsigned short* wWhhB   = (unsigned short*)alloc((size_t)H3_ * H_  * 2);
    unsigned short* wDecWih = (unsigned short*)alloc((size_t)H3_ * H3_ * 2);
    unsigned short* wDecWhh = (unsigned short*)alloc((size_t)H3_ * H_  * 2);
    unsigned short* wW1     = (unsigned short*)alloc((size_t)H_  * H3_ * 2);
    unsigned short* wOutW   = (unsigned short*)alloc((size_t)V_  * H_  * 2);
    // activations
    unsigned short* xbf     = (unsigned short*)alloc((size_t)S_ * B_ * E_ * 2);
    unsigned short* tgtbf   = (unsigned short*)alloc((size_t)S_ * B_ * E_ * 2);
    unsigned short* diobf   = (unsigned short*)alloc((size_t)E_ * 2);
    float*          encf32  = (float*)         alloc((size_t)S_ * B_ * H2_ * 4);
    unsigned short* encbf   = (unsigned short*)alloc((size_t)S_ * B_ * H2_ * 2);
    float*          GIf     = (float*)         alloc((size_t)S_ * B_ * H3_ * 4);
    float*          GIb     = (float*)         alloc((size_t)S_ * B_ * H3_ * 4);
    float*          GHf     = (float*)         alloc((size_t)B_ * H3_ * 4);
    float*          GHb     = (float*)         alloc((size_t)B_ * H3_ * 4);
    float*          hFf     = (float*)         alloc((size_t)B_ * H_ * 4);
    unsigned short* hFb     = (unsigned short*)alloc((size_t)B_ * H_ * 2);
    float*          hBf     = (float*)         alloc((size_t)B_ * H_ * 4);
    unsigned short* hBb     = (unsigned short*)alloc((size_t)B_ * H_ * 2);
    float*          hDf     = (float*)         alloc((size_t)B_ * H_ * 4);
    unsigned short* hDb     = (unsigned short*)alloc((size_t)B_ * H_ * 2);
    float*          EncProj = (float*)         alloc((size_t)S_ * B_ * H_ * 4);
    float*          HProj   = (float*)         alloc((size_t)B_ * H_ * 4);
    float*          scores  = (float*)         alloc((size_t)S_ * B_ * 4);
    float*          attw    = (float*)         alloc((size_t)S_ * B_ * 4);
    unsigned short* decin   = (unsigned short*)alloc((size_t)B_ * H3_ * 2);
    float*          decGI   = (float*)         alloc((size_t)B_ * H3_ * 4);
    float*          decGH   = (float*)         alloc((size_t)B_ * H3_ * 4);
    float*          logits  = (float*)         alloc((size_t)B_ * V_ * 4);
    (void)ws_size; (void)in_sizes; (void)n_in; (void)out_size;

    auto cvt = [&](const float* s, unsigned short* d, long n) {
        k_f32_to_bf16<<<(unsigned)((n + 255) / 256), 256, 0, stream>>>(s, d, n);
    };
    auto gemm = [&](const unsigned short* A, int lda, const unsigned short* W, int ldw,
                    const float* bias, float* o, int ldo, int M, int N, int K) {
        int blk = (M >> 5) * (N >> 6);         // 32x64 blocks, one wave each
        k_gemm_bf16<<<(blk + 7) / 8, 256, 0, stream>>>(A, lda, W, ldw, bias, o, ldo, M, N, K);
    };

    // ---- one-time conversions ----
    cvt(encWihF, wWihF,   (long)H3_ * E_);
    cvt(encWhhF, wWhhF,   (long)H3_ * H_);
    cvt(encWihB, wWihB,   (long)H3_ * E_);
    cvt(encWhhB, wWhhB,   (long)H3_ * H_);
    cvt(decWih,  wDecWih, (long)H3_ * H3_);
    cvt(decWhh,  wDecWhh, (long)H3_ * H_);
    cvt(attnW1,  wW1,     (long)H_  * H3_);
    cvt(outW,    wOutW,   (long)V_  * H_);
    cvt(decInitOut, diobf, E_);

    long nemb = (long)S_ * B_ * E_;
    k_embed<<<(unsigned)((nemb + 255) / 256), 256, 0, stream>>>(emb, input_seq,  xbf,   nemb);
    k_embed<<<(unsigned)((nemb + 255) / 256), 256, 0, stream>>>(emb, target_seq, tgtbf, nemb);

    int nBH = B_ * H_;
    k_bcast<<<(nBH + 255) / 256, 256, 0, stream>>>(encInit,      hFf, hFb, nBH);
    k_bcast<<<(nBH + 255) / 256, 256, 0, stream>>>(encInit + H_, hBf, hBb, nBH);
    k_bcast<<<(nBH + 255) / 256, 256, 0, stream>>>(decInitH,     hDf, hDb, nBH);

    // ---- hoisted input projections: GI = X @ Wih^T + bih, all steps at once
    gemm(xbf, E_, wWihF, E_, encBihF, GIf, H3_, S_ * B_, H3_, E_);
    gemm(xbf, E_, wWihB, E_, encBihB, GIb, H3_, S_ * B_, H3_, E_);

    // ---- bidirectional encoder recurrence ----
    for (int t = 0; t < S_; ++t) {
        int tb = S_ - 1 - t;
        gemm(hFb, H_, wWhhF, H_, encBhhF, GHf, H3_, B_, H3_, H_);
        k_gru_gate<<<(nBH + 255) / 256, 256, 0, stream>>>(
            GIf + (long)t * B_ * H3_, GHf, hFf, hFb,
            encf32 + (long)t * B_ * H2_, encbf + (long)t * B_ * H2_, nBH);
        gemm(hBb, H_, wWhhB, H_, encBhhB, GHb, H3_, B_, H3_, H_);
        k_gru_gate<<<(nBH + 255) / 256, 256, 0, stream>>>(
            GIb + (long)tb * B_ * H3_, GHb, hBf, hBb,
            encf32 + (long)tb * B_ * H2_ + H_, encbf + (long)tb * B_ * H2_ + H_, nBH);
    }

    // ---- hoisted attention encoder projection: EncProj = enc @ W1a^T ----
    gemm(encbf, H2_, wW1, H3_, nullptr, EncProj, H_, S_ * B_, H_, H2_);

    // ---- decoder ----
    int nBE = B_ * E_, nB2H = B_ * H2_;
    for (int t = 0; t < S_; ++t) {
        // HProj = h @ W1b^T   (W1b = cols [2H,3H) of attn_W1)
        gemm(hDb, H_, wW1 + H2_, H3_, nullptr, HProj, H_, B_, H_, H_);
        k_scores<<<(S_ * B_) / 8, 256, 0, stream>>>(EncProj, HProj, attnW2, scores);
        k_softmax_b<<<S_, 32, 0, stream>>>(scores, attw);
        k_glimpse<<<(nB2H + 255) / 256, 256, 0, stream>>>(encf32, attw, decin);
        k_decin_tok<<<(nBE + 255) / 256, 256, 0, stream>>>(
            xbf + (long)t * B_ * E_,
            (t == 0) ? diobf : (tgtbf + (long)(t - 1) * B_ * E_),
            (t == 0) ? 1 : 0, decin);
        gemm(decin, H3_, wDecWih, H3_, decBih, decGI, H3_, B_, H3_, H3_);
        gemm(hDb,   H_,  wDecWhh, H_,  decBhh, decGH, H3_, B_, H3_, H_);
        k_gru_gate<<<(nBH + 255) / 256, 256, 0, stream>>>(
            decGI, decGH, hDf, hDb, nullptr, nullptr, nBH);
        gemm(hDb, H_, wOutW, H_, outB, logits, V_, B_, V_, H_);
        k_softmax_v<<<B_, 256, 0, stream>>>(logits, out + (long)t * B_ * V_);
    }
}
